// Block_69423851372828
// MI455X (gfx1250) — matrix operand order repaired and checked by reading, not compiled, not run on hardware
//
#include <hip/hip_runtime.h>

typedef __attribute__((ext_vector_type(16))) __bf16 v16bf;
typedef __attribute__((ext_vector_type(8)))  __bf16 v8bf;
typedef __attribute__((ext_vector_type(8)))  float  v8f;

#define DEV __device__ __forceinline__

DEV unsigned short f2bf(float f) {
  unsigned u = __float_as_uint(f);
  unsigned r = u + 0x7FFFu + ((u >> 16) & 1u);   // round-to-nearest-even
  return (unsigned short)(r >> 16);
}

DEV v8f wmma_bf16(v16bf a, v16bf b, v8f c) {
  return __builtin_amdgcn_wmma_f32_16x16x32_bf16(false, a, false, b, (short)0, c, false, false);
}

// A-fragment (16x32 bf16): lane m=lane&15 holds row m; slots 0..7 = K base,
// slots 8..15 = K base+8. Two contiguous 16B groups.
DEV v16bf load_a_frag(const unsigned short* p0) {
  v8bf lo = *(const v8bf*)p0;
  v8bf hi = *(const v8bf*)(p0 + 8);
  v16bf r;
#pragma unroll
  for (int i = 0; i < 8; ++i) { r[i] = lo[i]; r[i + 8] = hi[i]; }
  return r;
}

// ---- CDNA5 async global->LDS copy (per-lane 16B), tracked by ASYNCcnt --------
DEV void async_b128(const void* gptr, void* lptr) {
  unsigned lds = (unsigned)(unsigned long long)lptr;  // low 32 bits = LDS offset
  asm volatile("global_load_async_to_lds_b128 %0, %1, off"
               :: "v"(lds), "v"(gptr) : "memory");
}
DEV void wait_async_le8() { asm volatile("s_wait_asynccnt 0x8" ::: "memory"); }
DEV void wait_async_le0() { asm volatile("s_wait_asynccnt 0x0" ::: "memory"); }

// ---------------- weight transpose + bf16 cast: out[n*K+k] = in[k*N+n] --------
__global__ __launch_bounds__(256) void wtrans_k(const float* __restrict__ in,
                                                unsigned short* __restrict__ out,
                                                int K, int N) {
  int idx = blockIdx.x * 256 + threadIdx.x;
  if (idx >= K * N) return;
  int k = idx / N, n = idx % N;
  out[(size_t)n * K + k] = f2bf(in[idx]);
}

// ---------------- LayerNorm (D=512) -> bf16 ----------------------------------
__global__ __launch_bounds__(256) void ln_bf16_k(const float* __restrict__ x,
                                                 const float* __restrict__ g,
                                                 const float* __restrict__ b,
                                                 unsigned short* __restrict__ out) {
  const int row = blockIdx.x, tid = threadIdx.x;
  const float* xr = x + (size_t)row * 512;
  float v0 = xr[tid], v1 = xr[tid + 256];
  float s = v0 + v1, s2 = v0 * v0 + v1 * v1;
#pragma unroll
  for (int m = 16; m >= 1; m >>= 1) {
    s  += __shfl_xor(s,  m, 32);
    s2 += __shfl_xor(s2, m, 32);
  }
  __shared__ float red[2][8];
  const int w = tid >> 5;
  if ((tid & 31) == 0) { red[0][w] = s; red[1][w] = s2; }
  __syncthreads();
  float ts = 0.f, ts2 = 0.f;
#pragma unroll
  for (int i = 0; i < 8; ++i) { ts += red[0][i]; ts2 += red[1][i]; }
  const float mu  = ts * (1.f / 512.f);
  const float var = ts2 * (1.f / 512.f) - mu * mu;
  const float rs  = rsqrtf(var + 1e-6f);
  out[(size_t)row * 512 + tid]       = f2bf((v0 - mu) * rs * g[tid]       + b[tid]);
  out[(size_t)row * 512 + tid + 256] = f2bf((v1 - mu) * rs * g[tid + 256] + b[tid + 256]);
}

// ---------------- WMMA GEMM with async double-buffered LDS staging -----------
// C[M,N] = A[M,K](bf16) * BT[N,K](bf16). 256 threads = 8 waves (2x4).
// Block tile 128x128, K-step 64 (16 WMMA per barrier pair). Wave tile 64x32.
// EPI 0: store bf16 [M][N]
// EPI 1: store bf16 V transposed [n][h][dk][s]   (N==512)
// EPI 2: outf = resid + C + bias  (fp32)
// EPI 3: store bf16 gelu(C + bias)
template <int EPI>
__global__ __launch_bounds__(256) void gemm_wmma_k(
    const unsigned short* __restrict__ A, const unsigned short* __restrict__ BT,
    const float* __restrict__ bias, const float* __restrict__ resid,
    unsigned short* __restrict__ outb, float* __restrict__ outf,
    int M, int N, int K) {
  const int tid = threadIdx.x, lane = tid & 31, w = tid >> 5;
  const int hl = lane >> 4, l15 = lane & 15;
  const int wm = (w >> 2) * 64, wn = (w & 3) * 32;
  const int tM = blockIdx.y * 128, tN = blockIdx.x * 128;

  __shared__ __align__(128) unsigned short lA[2][128 * 64];  // 16KB x2
  __shared__ __align__(128) unsigned short lB[2][128 * 64];  // 16KB x2

  v8f acc[4][2];
#pragma unroll
  for (int mt = 0; mt < 4; ++mt)
#pragma unroll
    for (int nt = 0; nt < 2; ++nt)
#pragma unroll
      for (int r = 0; r < 8; ++r) acc[mt][nt][r] = 0.f;

  // 128x64 tile = 1024 x 16B chunks; each thread copies 4 per tile (8 async/step)
  auto stage = [&](int buf, int kc) {
#pragma unroll
    for (int i = 0; i < 4; ++i) {
      const int ch = tid + i * 256;
      const int row = ch >> 3, co = (ch & 7) * 8;
      async_b128(A + (size_t)(tM + row) * K + kc + co, &lA[buf][row * 64 + co]);
    }
#pragma unroll
    for (int i = 0; i < 4; ++i) {
      const int ch = tid + i * 256;
      const int row = ch >> 3, co = (ch & 7) * 8;
      async_b128(BT + (size_t)(tN + row) * K + kc + co, &lB[buf][row * 64 + co]);
    }
  };

  stage(0, 0);
  const int nsteps = K >> 6;
  for (int s = 0; s < nsteps; ++s) {
    const int cur = s & 1;
    if (s + 1 < nsteps) { stage(cur ^ 1, (s + 1) * 64); wait_async_le8(); }
    else                { wait_async_le0(); }
    __syncthreads();

#pragma unroll
    for (int kk = 0; kk < 64; kk += 32) {
      v16bf a[4], b[2];
#pragma unroll
      for (int mt = 0; mt < 4; ++mt)
        a[mt] = load_a_frag(&lA[cur][(wm + mt * 16 + l15) * 64 + kk + hl * 16]);
#pragma unroll
      for (int nt = 0; nt < 2; ++nt)
        b[nt] = *(const v16bf*)&lB[cur][(wn + nt * 16 + l15) * 64 + kk + hl * 16];
#pragma unroll
      for (int mt = 0; mt < 4; ++mt)
#pragma unroll
        for (int nt = 0; nt < 2; ++nt)
          acc[mt][nt] = wmma_bf16(a[mt], b[nt], acc[mt][nt]);
    }
    __syncthreads();  // protect buffer about to be overwritten by next stage()
  }

#pragma unroll
  for (int mt = 0; mt < 4; ++mt)
#pragma unroll
    for (int nt = 0; nt < 2; ++nt)
#pragma unroll
      for (int r = 0; r < 8; ++r) {
        const int row = tM + wm + mt * 16 + r + hl * 8;
        const int col = tN + wn + nt * 16 + l15;
        float v = acc[mt][nt][r];
        if (EPI == 0) {
          outb[(size_t)row * N + col] = f2bf(v);
        } else if (EPI == 1) {
          const int nb = row >> 8, s = row & 255;
          const int hh = col >> 7, dk = col & 127;
          outb[((((size_t)nb * 4 + hh) * 128 + dk) << 8) + s] = f2bf(v);
        } else if (EPI == 2) {
          const size_t i = (size_t)row * N + col;
          outf[i] = resid[i] + v + bias[col];
        } else {
          float t = v + bias[col];
          float c = 0.7978845608028654f * (t + 0.044715f * t * t * t);
          outb[(size_t)row * N + col] = f2bf(0.5f * t * (1.f + tanhf(c)));
        }
      }
}

// ---------------- tri-block-diagonal flash attention --------------------------
// grid = N*H workgroups; 512 threads = 16 waves; wave w owns query rows w*16..+16.
// Online softmax over 6 KV chunks of 128 (3 neighbor blocks x 2 half-blocks).
// K chunk [kv][dk] and V chunk [dk][kv] are async-staged into double-buffered
// LDS (next chunk's 8 async ops overlap the current chunk's WMMA/softmax).
__global__ __launch_bounds__(512) void attn_k(
    const unsigned short* __restrict__ qb, const unsigned short* __restrict__ kb,
    const unsigned short* __restrict__ vt, const unsigned char* __restrict__ mask,
    unsigned short* __restrict__ outb) {
  const int NB = 128, S = 256, D = 512, H = 4, DK = 128;
  const int n = blockIdx.x >> 2, h = blockIdx.x & 3;
  const int tid = threadIdx.x, lane = tid & 31, w = tid >> 5;
  const int hl = lane >> 4, l15 = lane & 15;

  __shared__ __align__(128) unsigned short psc[16][512];      // 1KB P-scratch/wave
  __shared__ __align__(128) unsigned short kT[2][128 * 128];  // 32KB x2 [kv][dk]
  __shared__ __align__(128) unsigned short vT[2][128 * 128];  // 32KB x2 [dk][kv]
  unsigned short* pw = &psc[w][0];

  // Q A-fragments (DK=128 -> 4 K-chunks of 32)
  v16bf qf[4];
  const unsigned short* qp = qb + (size_t)(n * S + w * 16 + l15) * D + h * DK;
#pragma unroll
  for (int kc = 0; kc < 4; ++kc) qf[kc] = load_a_frag(qp + kc * 32 + hl * 16);

  float mrun[8], lrun[8];
  v8f oacc[8];
#pragma unroll
  for (int r = 0; r < 8; ++r) { mrun[r] = -1e30f; lrun[r] = 0.f; }
#pragma unroll
  for (int nt = 0; nt < 8; ++nt)
#pragma unroll
    for (int r = 0; r < 8; ++r) oacc[nt][r] = 0.f;

  const float scale = 0.08838834764831845f;  // DK^-0.5

  auto validj = [&](int jj) {
    const int b = n + (jj >> 1) - 1;
    return (b >= 0) && (b < NB);
  };
  auto stageKV = [&](int buf, int jj) {  // 8 async b128 per thread
    const int bn2 = n + (jj >> 1) - 1;
    const int c2 = jj & 1;
#pragma unroll
    for (int i = 0; i < 4; ++i) {
      const int ch = tid + i * 512;
      const int row = ch >> 4, co = (ch & 15) * 8;
      async_b128(kb + (size_t)(bn2 * S + c2 * 128 + row) * D + h * DK + co,
                 &kT[buf][row * 128 + co]);
    }
#pragma unroll
    for (int i = 0; i < 4; ++i) {
      const int ch = tid + i * 512;
      const int row = ch >> 4, co = (ch & 15) * 8;
      async_b128(vt + ((size_t)(bn2 * H + h) * DK + row) * S + c2 * 128 + co,
                 &vT[buf][row * 128 + co]);
    }
  };

  if (validj(0)) stageKV(0, 0);

  for (int j6 = 0; j6 < 6; ++j6) {
    const int off = (j6 >> 1) - 1;           // -1, 0, +1 neighbor block
    const int c = j6 & 1;                    // half-block (128 cols)
    const int bn = n + off;
    const bool valid = (bn >= 0) && (bn < NB);  // uniform across the block
    const int mi = (off == 0) ? 0 : (off == 1 ? 1 : 2);  // ld / lu / ll
    const int cur = j6 & 1;

    const bool pend = (j6 + 1 < 6) && validj(j6 + 1);
    if (pend) { stageKV(cur ^ 1, j6 + 1); wait_async_le8(); }
    else      { wait_async_le0(); }
    __syncthreads();

    v8f sacc[8];
#pragma unroll
    for (int ct = 0; ct < 8; ++ct)
#pragma unroll
      for (int r = 0; r < 8; ++r) sacc[ct][r] = 0.f;

    if (valid) {  // S = Q * K^T for this 16x128 strip (K from LDS)
#pragma unroll
      for (int ct = 0; ct < 8; ++ct) {
        const unsigned short* kp = &kT[cur][(ct * 16 + l15) * 128 + hl * 16];
        v16bf b0 = *(const v16bf*)(kp);
        v16bf b1 = *(const v16bf*)(kp + 32);
        v16bf b2 = *(const v16bf*)(kp + 64);
        v16bf b3 = *(const v16bf*)(kp + 96);
        sacc[ct] = wmma_bf16(qf[0], b0, sacc[ct]);
        sacc[ct] = wmma_bf16(qf[1], b1, sacc[ct]);
        sacc[ct] = wmma_bf16(qf[2], b2, sacc[ct]);
        sacc[ct] = wmma_bf16(qf[3], b3, sacc[ct]);
      }
    }

    // mask + scale (padded-out-of-range blocks keep score 0, matching reference)
    const unsigned char* mrow =
        mask + ((size_t)mi * NB + n) * S * S + (size_t)(w * 16 + hl * 8) * S + c * 128;
#pragma unroll
    for (int ct = 0; ct < 8; ++ct)
#pragma unroll
      for (int r = 0; r < 8; ++r) {
        const unsigned char mk = mrow[(size_t)r * S + ct * 16 + l15];
        sacc[ct][r] = mk ? sacc[ct][r] * scale : -1e30f;
      }

    // online softmax: row max (cross-lane within 16-lane half)
    float mnew[8];
#pragma unroll
    for (int r = 0; r < 8; ++r) {
      float mx = sacc[0][r];
#pragma unroll
      for (int ct = 1; ct < 8; ++ct) mx = fmaxf(mx, sacc[ct][r]);
      mx = fmaxf(mx, __shfl_xor(mx, 8, 32));
      mx = fmaxf(mx, __shfl_xor(mx, 4, 32));
      mx = fmaxf(mx, __shfl_xor(mx, 2, 32));
      mx = fmaxf(mx, __shfl_xor(mx, 1, 32));
      mnew[r] = fmaxf(mrun[r], mx);
    }
#pragma unroll
    for (int r = 0; r < 8; ++r) {
      const float f = __expf(mrun[r] - mnew[r]);
      lrun[r] *= f;
      mrun[r] = mnew[r];
#pragma unroll
      for (int nt = 0; nt < 8; ++nt) oacc[nt][r] *= f;
    }
    float rsum[8];
#pragma unroll
    for (int r = 0; r < 8; ++r) rsum[r] = 0.f;
#pragma unroll
    for (int ct = 0; ct < 8; ++ct)
#pragma unroll
      for (int r = 0; r < 8; ++r) {
        const float p = __expf(sacc[ct][r] - mrun[r]);
        sacc[ct][r] = p;
        rsum[r] += p;
      }
#pragma unroll
    for (int r = 0; r < 8; ++r) {
      float sm = rsum[r];
      sm += __shfl_xor(sm, 8, 32);
      sm += __shfl_xor(sm, 4, 32);
      sm += __shfl_xor(sm, 2, 32);
      sm += __shfl_xor(sm, 1, 32);
      lrun[r] += sm;
    }

    if (valid) {  // O += P * V  (V^T from LDS)
#pragma unroll
      for (int kc2 = 0; kc2 < 4; ++kc2) {
        // C-layout -> A-layout through per-wave LDS scratch (16x32 bf16)
#pragma unroll
        for (int r = 0; r < 8; ++r) {
          const int prow = r + hl * 8;
          pw[prow * 32 + l15]      = f2bf(sacc[2 * kc2][r]);
          pw[prow * 32 + 16 + l15] = f2bf(sacc[2 * kc2 + 1][r]);
        }
        asm volatile("s_wait_dscnt 0" ::: "memory");
        v16bf pf = load_a_frag(pw + l15 * 32 + hl * 16);
        const unsigned short* vp0 = &vT[cur][kc2 * 32 + hl * 16];
#pragma unroll
        for (int nt = 0; nt < 8; ++nt) {
          v16bf bv = *(const v16bf*)(vp0 + (nt * 16 + l15) * 128);
          oacc[nt] = wmma_bf16(pf, bv, oacc[nt]);
        }
      }
    }
    __syncthreads();  // buffer cur is overwritten by stage at iteration j6+1
  }

#pragma unroll
  for (int r = 0; r < 8; ++r) lrun[r] = 1.f / lrun[r];
  const size_t obase = (size_t)(n * S + w * 16);
#pragma unroll
  for (int nt = 0; nt < 8; ++nt)
#pragma unroll
    for (int r = 0; r < 8; ++r) {
      const size_t row = obase + r + hl * 8;
      const int col = h * DK + nt * 16 + l15;
      outb[row * D + col] = f2bf(oacc[nt][r] * lrun[r]);
    }
}

// -----------------------------------------------------------------------------
extern "C" void kernel_launch(void* const* d_in, const int* in_sizes, int n_in,
                              void* d_out, int out_size, void* d_ws, size_t ws_size,
                              hipStream_t stream) {
  (void)in_sizes; (void)n_in; (void)out_size; (void)ws_size;
  const int M = 32768, D = 512, FF = 2048, NB = 128, H = 4;

  const float* x            = (const float*)d_in[0];
  const unsigned char* mask = (const unsigned char*)d_in[1];
  const float* Wq  = (const float*)d_in[2];
  const float* Wk  = (const float*)d_in[3];
  const float* Wv  = (const float*)d_in[4];
  const float* Wo  = (const float*)d_in[5];
  const float* bo  = (const float*)d_in[6];
  const float* W1  = (const float*)d_in[7];
  const float* b1  = (const float*)d_in[8];
  const float* W2  = (const float*)d_in[9];
  const float* b2  = (const float*)d_in[10];
  const float* g1  = (const float*)d_in[11];
  const float* be1 = (const float*)d_in[12];
  const float* g2  = (const float*)d_in[13];
  const float* be2 = (const float*)d_in[14];

  char* ws = (char*)d_ws;
  size_t off = 0;
  auto alloc = [&](size_t bytes) -> void* {
    void* p = ws + off;
    off += (bytes + 255) & ~(size_t)255;
    return p;
  };
  unsigned short* hln  = (unsigned short*)alloc((size_t)M * D * 2);
  unsigned short* qbuf = (unsigned short*)alloc((size_t)M * D * 2);
  unsigned short* kbuf = (unsigned short*)alloc((size_t)M * D * 2);
  unsigned short* vtb  = (unsigned short*)alloc((size_t)M * D * 2);
  float*          y    = (float*)alloc((size_t)M * D * 4);
  unsigned short* gact = (unsigned short*)alloc((size_t)M * FF * 2);
  unsigned short* WqT  = (unsigned short*)alloc((size_t)D * D * 2);
  unsigned short* WkT  = (unsigned short*)alloc((size_t)D * D * 2);
  unsigned short* WvT  = (unsigned short*)alloc((size_t)D * D * 2);
  unsigned short* WoT  = (unsigned short*)alloc((size_t)D * D * 2);
  unsigned short* W1T  = (unsigned short*)alloc((size_t)D * FF * 2);
  unsigned short* W2T  = (unsigned short*)alloc((size_t)FF * D * 2);
  unsigned short* attnb = hln;   // hln dead after V projection
  unsigned short* h2    = qbuf;  // qbuf dead after attention

  wtrans_k<<<(D * D + 255) / 256, 256, 0, stream>>>(Wq, WqT, D, D);
  wtrans_k<<<(D * D + 255) / 256, 256, 0, stream>>>(Wk, WkT, D, D);
  wtrans_k<<<(D * D + 255) / 256, 256, 0, stream>>>(Wv, WvT, D, D);
  wtrans_k<<<(D * D + 255) / 256, 256, 0, stream>>>(Wo, WoT, D, D);
  wtrans_k<<<(D * FF + 255) / 256, 256, 0, stream>>>(W1, W1T, D, FF);
  wtrans_k<<<(FF * D + 255) / 256, 256, 0, stream>>>(W2, W2T, FF, D);

  ln_bf16_k<<<M, 256, 0, stream>>>(x, g1, be1, hln);

  dim3 blk(256);
  gemm_wmma_k<0><<<dim3(D / 128, M / 128), blk, 0, stream>>>(hln, WqT, nullptr, nullptr, qbuf, nullptr, M, D, D);
  gemm_wmma_k<0><<<dim3(D / 128, M / 128), blk, 0, stream>>>(hln, WkT, nullptr, nullptr, kbuf, nullptr, M, D, D);
  gemm_wmma_k<1><<<dim3(D / 128, M / 128), blk, 0, stream>>>(hln, WvT, nullptr, nullptr, vtb, nullptr, M, D, D);

  attn_k<<<NB * H, 512, 0, stream>>>(qbuf, kbuf, vtb, mask, attnb);

  gemm_wmma_k<2><<<dim3(D / 128, M / 128), blk, 0, stream>>>(attnb, WoT, bo, x, nullptr, y, M, D, D);
  ln_bf16_k<<<M, 256, 0, stream>>>(y, g2, be2, h2);
  gemm_wmma_k<3><<<dim3(FF / 128, M / 128), blk, 0, stream>>>(h2, W1T, b1, nullptr, gact, nullptr, M, FF, D);
  gemm_wmma_k<2><<<dim3(D / 128, M / 128), blk, 0, stream>>>(gact, W2T, b2, y, nullptr, (float*)d_out, M, D, FF);
}
